// MultiHeadSelfAttention_137438954025
// MI455X (gfx1250) — matrix operand order repaired and checked by reading, not compiled, not run on hardware
//
#include <hip/hip_runtime.h>
#include <hip/hip_bf16.h>

// MI455X / gfx1250 multi-head self-attention
//   B=2, N=4096, DIM=512, H=8, DH=64
// Phase 1: QKV projection GEMM (bf16 WMMA, f32 accum), q/v/k chunk order.
//          Q is pre-scaled by 1/sqrt(DH); V written transposed [bh][d][n].
// Phase 2: flash attention, 64-key chunks double-buffered via the Tensor
//          Data Mover (tensor_load_to_lds, TENSORcnt), bf16 WMMA for QK^T
//          and P*V, f32 online softmax with ds_swizzle reductions.

typedef __bf16 bf16_t;
typedef __bf16  v16bf __attribute__((ext_vector_type(16)));
typedef float   v8f   __attribute__((ext_vector_type(8)));
typedef unsigned int v4u __attribute__((ext_vector_type(4)));
typedef int          v8i __attribute__((ext_vector_type(8)));
typedef int          v4i __attribute__((ext_vector_type(4)));

#define BATCH 2
#define NSEQ  4096
#define DIMC  512
#define HEADS 8
#define DHEAD 64
#define NCOL  (3 * DIMC)
#define CHUNK 64            // keys per attention iteration

// ---------------------------------------------------------------------------
// helpers
// ---------------------------------------------------------------------------
__device__ __forceinline__ unsigned pack2bf(float lo, float hi) {
    unsigned short a = __builtin_bit_cast(unsigned short, (__bf16)lo);
    unsigned short b = __builtin_bit_cast(unsigned short, (__bf16)hi);
    return (unsigned)a | ((unsigned)b << 16);     // v_cvt_pk_bf16_f32
}

// ds_swizzle xor-lane reductions within 16-lane half groups (masks <= 8).
template <int PAT>
__device__ __forceinline__ float swz_xor(float v) {
    return __builtin_bit_cast(float,
        __builtin_amdgcn_ds_swizzle(__builtin_bit_cast(int, v), PAT));
}
#define SWZ_X1 ((1 << 10) | 0x1f)
#define SWZ_X2 ((2 << 10) | 0x1f)
#define SWZ_X4 ((4 << 10) | 0x1f)
#define SWZ_X8 ((8 << 10) | 0x1f)
__device__ __forceinline__ float redmax16(float v) {
    v = fmaxf(v, swz_xor<SWZ_X1>(v));
    v = fmaxf(v, swz_xor<SWZ_X2>(v));
    v = fmaxf(v, swz_xor<SWZ_X4>(v));
    v = fmaxf(v, swz_xor<SWZ_X8>(v));
    return v;
}
__device__ __forceinline__ float redsum16(float v) {
    v += swz_xor<SWZ_X1>(v);
    v += swz_xor<SWZ_X2>(v);
    v += swz_xor<SWZ_X4>(v);
    v += swz_xor<SWZ_X8>(v);
    return v;
}

#define LDS_OFF(p) ((unsigned)(unsigned long long)(p))

// ---------------------------------------------------------------------------
// Tensor Data Mover: load a 2D tile (rows x row_units of 8B) from global to
// LDS, packed contiguously. D# per CDNA5 ISA ch.8 (data_size=8B, type=2,
// huge tensor dims => no OOB clipping, no padding/iteration/multicast).
// ---------------------------------------------------------------------------
__device__ __forceinline__ void tdm_load_2d(unsigned lds_addr, const void* gptr,
                                            unsigned tile_d0 /*8B units*/,
                                            unsigned tile_d1 /*rows*/,
                                            unsigned long long stride0_units)
{
    const unsigned long long ga = (unsigned long long)gptr;
    const unsigned td0 = 0x40000000u, td1 = 0x40000000u;  // huge: no clip

    v4u g0;
    g0.x = 1u;                                       // count=1 valid, user D#
    g0.y = lds_addr;                                 // lds_addr [63:32]
    g0.z = (unsigned)(ga & 0xFFFFFFFFu);             // global_addr lo
    g0.w = (unsigned)((ga >> 32) & 0x01FFFFFFu)      // global_addr [56:32]
         | 0x80000000u;                              // type=2 ("image")

    v8i g1;
    g1[0] = (int)(3u << 16);                         // data_size=3 (8 bytes)
    g1[1] = (int)((td0 & 0xFFFFu) << 16);            // tensor_dim0 [79:48]
    g1[2] = (int)(((td0 >> 16) & 0xFFFFu) | ((td1 & 0xFFFFu) << 16));
    g1[3] = (int)(((td1 >> 16) & 0xFFFFu) | ((tile_d0 & 0xFFFFu) << 16));
    g1[4] = (int)(tile_d1 & 0xFFFFu);                // tile_dim1; tile_dim2=0
    g1[5] = (int)(unsigned)(stride0_units & 0xFFFFFFFFu);
    g1[6] = (int)(unsigned)((stride0_units >> 32) & 0xFFFFu);
    g1[7] = 0;

    v4i z4 = {0, 0, 0, 0};
#if defined(__clang_major__) && (__clang_major__ >= 23)
    v8i z8 = {0, 0, 0, 0, 0, 0, 0, 0};
    __builtin_amdgcn_tensor_load_to_lds(g0, g1, z4, z4, z8, 0);
#else
    __builtin_amdgcn_tensor_load_to_lds(g0, g1, z4, z4, 0);
#endif
}

// ---------------------------------------------------------------------------
// Kernel 1: qkv = x @ W + b -> bf16 Q*scale [bh][n][d], K [bh][n][d],
//           V^T [bh][d][n].  Block: 256 threads (8 waves), tile 128x64.
// ---------------------------------------------------------------------------
__global__ __launch_bounds__(256)
void qkv_gemm_kernel(const float* __restrict__ x, const float* __restrict__ W,
                     const float* __restrict__ bias,
                     bf16_t* __restrict__ qws, bf16_t* __restrict__ kws,
                     bf16_t* __restrict__ vtws)
{
    __shared__ __align__(16) bf16_t lA[128 * 32];   // [row][k]
    __shared__ __align__(16) bf16_t lB[64 * 32];    // [n][k] (transposed W)

    const int tid  = threadIdx.x;
    const int wave = tid >> 5;
    const int lane = tid & 31;
    const int hi   = lane >> 4;
    const int l16  = lane & 15;

    const int rowBase = blockIdx.x * 128;
    const int colBase = blockIdx.y * 64;

    const int bn  = tid & 63;          // W tile: n within tile
    const int bkq = (tid >> 6) << 3;   // W tile: k start (0,8,16,24)

    v8f acc[4] = {};

    for (int k0 = 0; k0 < DIMC; k0 += 32) {
        __syncthreads();
        #pragma unroll
        for (int i = 0; i < 4; ++i) {
            int e4 = tid + i * 256;
            int r  = e4 >> 3;
            int kq = (e4 & 7) << 2;
            const float4 f =
                *(const float4*)&x[(size_t)(rowBase + r) * DIMC + k0 + kq];
            uint2 u;
            u.x = pack2bf(f.x, f.y);
            u.y = pack2bf(f.z, f.w);
            *(uint2*)&lA[r * 32 + kq] = u;
        }
        {
            float v[8];
            #pragma unroll
            for (int j = 0; j < 8; ++j)
                v[j] = W[(size_t)(k0 + bkq + j) * NCOL + colBase + bn];
            uint4 u;
            u.x = pack2bf(v[0], v[1]);
            u.y = pack2bf(v[2], v[3]);
            u.z = pack2bf(v[4], v[5]);
            u.w = pack2bf(v[6], v[7]);
            *(uint4*)&lB[bn * 32 + bkq] = u;
        }
        __syncthreads();

        v16bf af;
        const int arow = wave * 16 + l16;
        #pragma unroll
        for (int vv = 0; vv < 8; ++vv) {
            int kk = ((vv < 4) ? 0 : 16) + (hi ? 8 : 0) + 2 * (vv & 3);
            af[2 * vv]     = lA[arow * 32 + kk];
            af[2 * vv + 1] = lA[arow * 32 + kk + 1];
        }
        #pragma unroll
        for (int s = 0; s < 4; ++s) {
            v16bf bfr;
            const int n = s * 16 + l16;
            #pragma unroll
            for (int vv = 0; vv < 8; ++vv) {
                int kk = ((vv < 4) ? 0 : 16) + (hi ? 8 : 0) + 2 * (vv & 3);
                bfr[2 * vv]     = lB[n * 32 + kk];
                bfr[2 * vv + 1] = lB[n * 32 + kk + 1];
            }
            acc[s] = __builtin_amdgcn_wmma_f32_16x16x32_bf16(
                false, af, false, bfr, (short)0, acc[s], false, false);
        }
    }

    // Epilogue: bias, q/v/k routing (chunk order q, v, k), Q pre-scaled.
    #pragma unroll
    for (int s = 0; s < 4; ++s) {
        const int col    = colBase + s * 16 + l16;
        const float bv   = bias[col];
        const int chunk  = col / DIMC;   // 0=q, 1=v, 2=k
        const int within = col % DIMC;
        const int h      = within / DHEAD;
        const int d      = within % DHEAD;
        const float postScale = (chunk == 0) ? 0.125f : 1.0f; // 1/sqrt(64)
        #pragma unroll
        for (int vv = 0; vv < 8; ++vv) {
            const int rr = rowBase + wave * 16 + vv + (hi ? 8 : 0);
            const int b  = rr >> 12;
            const int n  = rr & (NSEQ - 1);
            const bf16_t val = (bf16_t)((acc[s][vv] + bv) * postScale);
            const int bh = b * HEADS + h;
            if (chunk == 1) {
                vtws[((size_t)bh * DHEAD + d) * NSEQ + n] = val;   // V^T
            } else {
                bf16_t* dst = (chunk == 0) ? qws : kws;
                dst[((size_t)bh * NSEQ + n) * DHEAD + d] = val;
            }
        }
    }
}

// ---------------------------------------------------------------------------
// Kernel 2: flash attention. 128 threads (4 waves), each wave owns 16 query
// rows. 64-key chunks double-buffered via TDM tensor_load_to_lds.
// ---------------------------------------------------------------------------
__global__ __launch_bounds__(128)
void attn_kernel(const bf16_t* __restrict__ qws, const bf16_t* __restrict__ kws,
                 const bf16_t* __restrict__ vtws, float* __restrict__ out)
{
    __shared__ __align__(16) bf16_t lK[2][CHUNK * 64];   // [key][d]
    __shared__ __align__(16) bf16_t lVt[2][64 * CHUNK];  // [d][key]
    __shared__ __align__(16) bf16_t lP[4 * 16 * CHUNK];  // per-wave P scratch

    const int tid  = threadIdx.x;
    const int wave = tid >> 5;
    const int lane = tid & 31;
    const int hi   = lane >> 4;
    const int l16  = lane & 15;

    const int bh = blockIdx.y;
    const int qb = blockIdx.x * 64 + wave * 16;

    const bf16_t* qh  = qws  + (size_t)bh * NSEQ * DHEAD;
    const bf16_t* kh  = kws  + (size_t)bh * NSEQ * DHEAD;
    const bf16_t* vth = vtws + (size_t)bh * DHEAD * NSEQ;

    const unsigned lKo[2]  = { LDS_OFF(&lK[0][0]),  LDS_OFF(&lK[1][0]) };
    const unsigned lVto[2] = { LDS_OFF(&lVt[0][0]), LDS_OFF(&lVt[1][0]) };

    // Preload Q A-fragments (d 0..31, 32..63); Q already holds 1/sqrt(dh).
    v16bf afQ[2];
    {
        const int qrow = qb + l16;
        #pragma unroll
        for (int ks = 0; ks < 2; ++ks) {
            #pragma unroll
            for (int vv = 0; vv < 8; ++vv) {
                int dd = ks * 32 + ((vv < 4) ? 0 : 16) + (hi ? 8 : 0) + 2 * (vv & 3);
                afQ[ks][2 * vv]     = qh[(size_t)qrow * DHEAD + dd];
                afQ[ks][2 * vv + 1] = qh[(size_t)qrow * DHEAD + dd + 1];
            }
        }
    }

    // TDM stage of one 64-key chunk (wave 0 issues; EXEC ignored by TDM).
    auto issue_chunk = [&](int kc, int buf) {
        // K: contiguous 64x64 bf16 = 8KB = 1024 x 8B, single "row"
        tdm_load_2d(lKo[buf], kh + (size_t)kc * DHEAD, 1024u, 1u, 1024ull);
        // V^T: 64 rows (d), each 64 keys = 128B = 16 x 8B; row stride NSEQ
        tdm_load_2d(lVto[buf], vth + kc, 16u, 64u,
                    (unsigned long long)(NSEQ * 2 / 8));
    };

    float mrow[8], lrow[8];
    #pragma unroll
    for (int vv = 0; vv < 8; ++vv) { mrow[vv] = -1.0e30f; lrow[vv] = 0.0f; }
    v8f accO[4] = {};

    if (wave == 0) issue_chunk(0, 0);

    for (int kc = 0; kc < NSEQ; kc += CHUNK) {
        const int buf = (kc / CHUNK) & 1;
        __builtin_amdgcn_s_wait_tensorcnt(0);
        __syncthreads();
        if (wave == 0 && kc + CHUNK < NSEQ) issue_chunk(kc + CHUNK, buf ^ 1);

        const bf16_t* cK  = &lK[buf][0];
        const bf16_t* cVt = &lVt[buf][0];

        // S = Q K^T: 4 key subtiles x 2 k-steps = 8 WMMAs
        v8f st[4] = {};
        #pragma unroll
        for (int ks = 0; ks < 2; ++ks) {
            #pragma unroll
            for (int c = 0; c < 4; ++c) {
                v16bf bk;
                #pragma unroll
                for (int vv = 0; vv < 8; ++vv) {
                    int dd = ks * 32 + ((vv < 4) ? 0 : 16) + (hi ? 8 : 0) + 2 * (vv & 3);
                    bk[2 * vv]     = cK[(c * 16 + l16) * 64 + dd];
                    bk[2 * vv + 1] = cK[(c * 16 + l16) * 64 + dd + 1];
                }
                st[c] = __builtin_amdgcn_wmma_f32_16x16x32_bf16(
                    false, afQ[ks], false, bk, (short)0, st[c], false, false);
            }
        }

        // Online softmax per row (rows live in 16-lane half groups)
        #pragma unroll
        for (int vv = 0; vv < 8; ++vv) {
            float S0 = st[0][vv], S1 = st[1][vv], S2 = st[2][vv], S3 = st[3][vv];
            float mc = redmax16(fmaxf(fmaxf(S0, S1), fmaxf(S2, S3)));
            float mnew = fmaxf(mrow[vv], mc);
            float sc   = __expf(mrow[vv] - mnew);
            float p0 = __expf(S0 - mnew), p1 = __expf(S1 - mnew);
            float p2 = __expf(S2 - mnew), p3 = __expf(S3 - mnew);
            float rs = redsum16((p0 + p1) + (p2 + p3));
            lrow[vv] = lrow[vv] * sc + rs;
            mrow[vv] = mnew;
            #pragma unroll
            for (int s = 0; s < 4; ++s) accO[s][vv] *= sc;
            const int r = vv + (hi ? 8 : 0);
            lP[wave * (16 * CHUNK) + r * CHUNK +  0 + l16] = (bf16_t)p0;
            lP[wave * (16 * CHUNK) + r * CHUNK + 16 + l16] = (bf16_t)p1;
            lP[wave * (16 * CHUNK) + r * CHUNK + 32 + l16] = (bf16_t)p2;
            lP[wave * (16 * CHUNK) + r * CHUNK + 48 + l16] = (bf16_t)p3;
        }
        asm volatile("" ::: "memory");   // wave-private LDS: order store->load

        // P back as two A-fragments (16x32 each); O += P * V : 8 WMMAs
        v16bf pf[2];
        #pragma unroll
        for (int ks2 = 0; ks2 < 2; ++ks2) {
            #pragma unroll
            for (int vv = 0; vv < 8; ++vv) {
                int kk = ks2 * 32 + ((vv < 4) ? 0 : 16) + (hi ? 8 : 0) + 2 * (vv & 3);
                pf[ks2][2 * vv]     = lP[wave * (16 * CHUNK) + l16 * CHUNK + kk];
                pf[ks2][2 * vv + 1] = lP[wave * (16 * CHUNK) + l16 * CHUNK + kk + 1];
            }
        }
        #pragma unroll
        for (int s = 0; s < 4; ++s) {
            #pragma unroll
            for (int ks2 = 0; ks2 < 2; ++ks2) {
                v16bf bv;
                const int n = s * 16 + l16;                 // d column
                #pragma unroll
                for (int vv = 0; vv < 8; ++vv) {
                    int kk = ks2 * 32 + ((vv < 4) ? 0 : 16) + (hi ? 8 : 0) + 2 * (vv & 3);  // key index
                    bv[2 * vv]     = cVt[n * CHUNK + kk];
                    bv[2 * vv + 1] = cVt[n * CHUNK + kk + 1];
                }
                accO[s] = __builtin_amdgcn_wmma_f32_16x16x32_bf16(
                    false, pf[ks2], false, bv, (short)0, accO[s], false, false);
            }
        }
        __syncthreads();   // compute done before next overwrite of this buf
    }

    // Epilogue: normalize by l, write fp32 out[b][n][h*64+d]
    const int b = bh >> 3;
    const int h = bh & 7;
    #pragma unroll
    for (int vv = 0; vv < 8; ++vv) {
        const float rinv = 1.0f / lrow[vv];
        const int n = qb + vv + (hi ? 8 : 0);
        #pragma unroll
        for (int s = 0; s < 4; ++s) {
            const int d = s * 16 + l16;
            out[((size_t)(b * NSEQ + n)) * DIMC + h * DHEAD + d] =
                accO[s][vv] * rinv;
        }
    }
}

// ---------------------------------------------------------------------------
extern "C" void kernel_launch(void* const* d_in, const int* in_sizes, int n_in,
                              void* d_out, int out_size, void* d_ws, size_t ws_size,
                              hipStream_t stream)
{
    const float* x    = (const float*)d_in[0];   // [2,4096,512]
    const float* W    = (const float*)d_in[1];   // [512,1536]
    const float* bias = (const float*)d_in[2];   // [1536]
    float* out        = (float*)d_out;           // [2,4096,512]

    const size_t headElems = (size_t)BATCH * HEADS * NSEQ * DHEAD; // 4M elems
    bf16_t* qws  = (bf16_t*)d_ws;
    bf16_t* kws  = qws + headElems;
    bf16_t* vtws = kws + headElems;              // transposed V

    dim3 g1((BATCH * NSEQ) / 128, NCOL / 64);    // 64 x 24
    qkv_gemm_kernel<<<g1, 256, 0, stream>>>(x, W, bias, qws, kws, vtws);

    dim3 g2(NSEQ / 64, BATCH * HEADS);           // 64 x 16
    attn_kernel<<<g2, 128, 0, stream>>>(qws, kws, vtws, out);
}
